// AttentionBlock_49074296324680
// MI455X (gfx1250) — matrix operand order repaired and checked by reading, not compiled, not run on hardware
//
#include <hip/hip_runtime.h>

// ---------------------------------------------------------------------------
// AttentionBlock for MI455X (gfx1250, wave32, WMMA f32_16x16x32_f16)
//   x:(4,256,64,64) f32 -> GN(32) -> qkv(768x256) -> 4-head attn (L=4096,d=64)
//   -> proj(256x256) -> +residual
// f16 operands, fp32 accumulation, fp32 online softmax (flash attention).
// Attention: 8 waves/block share double-buffered K/V tiles in LDS, staged via
// GLOBAL_LOAD_ASYNC_TO_LDS (ASYNCcnt) when the builtin is available.
// GEMMs: 64x64 register-blocked tiles per wave (4x4 accumulators, 16 WMMA/k).
// ---------------------------------------------------------------------------

typedef __attribute__((ext_vector_type(16))) _Float16 v16h;
typedef __attribute__((ext_vector_type(8)))  _Float16 v8h;
typedef __attribute__((ext_vector_type(8)))  float    v8f;
typedef __attribute__((ext_vector_type(4)))  int      v4i;

#define B_TOT 4
#define C_TOT 256
#define L_TOT 4096
#define NH    4
#define CH    64
#define BH_TOT (B_TOT * NH)   // 16
#define QKV_SCALE 0.35355339059327373f  // 1/sqrt(sqrt(64))

// ---- async global->LDS staging (CDNA5 ASYNCcnt path), with safe fallback ----
#if defined(__has_builtin)
#  if __has_builtin(__builtin_amdgcn_global_load_async_to_lds_b128)
#    define HAVE_ASYNC_LDS 1
#  endif
#endif
#ifndef HAVE_ASYNC_LDS
#  define HAVE_ASYNC_LDS 0
#endif

#if HAVE_ASYNC_LDS
__device__ __forceinline__ void async_copy16(const _Float16* g, _Float16* l) {
  __builtin_amdgcn_global_load_async_to_lds_b128(
      (__attribute__((address_space(1))) v4i*)g,
      (__attribute__((address_space(3))) v4i*)l, 0, 0);
}
__device__ __forceinline__ void wait_async0() {
#  if __has_builtin(__builtin_amdgcn_s_wait_asynccnt)
  __builtin_amdgcn_s_wait_asynccnt(0);
#  else
  asm volatile("s_wait_asynccnt 0" ::: "memory");
#  endif
}
#endif

__device__ __forceinline__ v16h combine16(v8h lo, v8h hi) {
  return __builtin_shufflevector(lo, hi, 0,1,2,3,4,5,6,7,8,9,10,11,12,13,14,15);
}

// A fragment (16x32 f16): lane m = lane&15, hi = lane>>4.
// halves 0..7  -> K = hi*8 + j ; halves 8..15 -> K = 16 + hi*8 + j
__device__ __forceinline__ v16h load_a_frag(const _Float16* rowPtr, int hi) {
  v8h lo = *(const v8h*)(rowPtr + hi * 8);
  v8h hh = *(const v8h*)(rowPtr + 16 + hi * 8);
  return combine16(lo, hh);
}

// B fragment (32x16 f16): lane n = lane&15, hi = lane>>4.
// halves 0..7  -> K = hi*8 + j ; halves 8..15 -> K = 16 + hi*8 + j
__device__ __forceinline__ v16h load_b_frag(const _Float16* colPtr, int hi) {
  v8h lo = *(const v8h*)(colPtr + hi * 8);
  v8h hh = *(const v8h*)(colPtr + 16 + hi * 8);
  return combine16(lo, hh);
}

__device__ __forceinline__ v8f wmma_f16(v16h a, v16h b, v8f c) {
  return __builtin_amdgcn_wmma_f32_16x16x32_f16(false, a, false, b, (short)0, c, false, false);
}

// ---------------------------------------------------------------- weights f32->f16
__global__ void f32_to_f16_kernel(const float* __restrict__ in, _Float16* __restrict__ out, int n) {
  int i = blockIdx.x * 256 + threadIdx.x;
  if (i < n) out[i] = (_Float16)in[i];
}

// ---------------------------------------------------------------- GroupNorm stats
__global__ void gn_stats_kernel(const float* __restrict__ x,
                                float* __restrict__ mean, float* __restrict__ rstd) {
  const int bg = blockIdx.x;         // b*32+g, memory-contiguous groups of 32768
  const int tid = threadIdx.x;       // 256 threads
  const float* p = x + (size_t)bg * 32768;
  float s = 0.f, ss = 0.f;
  for (int i = tid; i < 32768; i += 256) { float v = p[i]; s += v; ss += v * v; }
  __shared__ float sh_s[256], sh_q[256];
  sh_s[tid] = s; sh_q[tid] = ss;
  __syncthreads();
  for (int off = 128; off > 0; off >>= 1) {
    if (tid < off) { sh_s[tid] += sh_s[tid + off]; sh_q[tid] += sh_q[tid + off]; }
    __syncthreads();
  }
  if (tid == 0) {
    float m = sh_s[0] * (1.f / 32768.f);
    float var = sh_q[0] * (1.f / 32768.f) - m * m;
    mean[bg] = m;
    rstd[bg] = rsqrtf(var + 1e-5f);
  }
}

// ------------------------------------------------- GN apply + transpose to (B,L,C) f16
__global__ void gn_apply_t_kernel(const float* __restrict__ x,
                                  const float* __restrict__ gw, const float* __restrict__ gb,
                                  const float* __restrict__ mean, const float* __restrict__ rstd,
                                  _Float16* __restrict__ xnT) {
  __shared__ _Float16 tile[32][33];
  const int b = blockIdx.z, c0 = blockIdx.y * 32, l0 = blockIdx.x * 32;
  const int tx = threadIdx.x, ty = threadIdx.y;
#pragma unroll
  for (int j = 0; j < 4; ++j) {
    int c = c0 + ty + j * 8;
    int l = l0 + tx;
    int g = c >> 3;
    float m = mean[b * 32 + g], rs = rstd[b * 32 + g];
    float val = (x[((size_t)(b * C_TOT + c)) * L_TOT + l] - m) * rs * gw[c] + gb[c];
    tile[ty + j * 8][tx] = (_Float16)val;
  }
  __syncthreads();
#pragma unroll
  for (int j = 0; j < 4; ++j) {
    int l = l0 + ty + j * 8;
    xnT[((size_t)b * L_TOT + l) * C_TOT + c0 + tx] = tile[tx][ty + j * 8];
  }
}

// ---------------------------------------------------------------- QKV GEMM (WMMA)
// 64x64 tile per wave, 4x4 v8f accumulators, K=256.
// q,k stored transposed (BH,L,64) pre-scaled; v stored (BH,64,L).
__global__ __launch_bounds__(128) void qkv_gemm_kernel(
    const _Float16* __restrict__ W,      // 768x256
    const _Float16* __restrict__ xnT,    // (B,L,256)
    const float* __restrict__ bias,      // 768
    _Float16* __restrict__ qT, _Float16* __restrict__ kT,  // (BH,L,64)
    _Float16* __restrict__ vmat)                           // (BH,64,L)
{
  const int wave = threadIdx.x >> 5, lane = threadIdx.x & 31;
  const int n = lane & 15, hi = lane >> 4;
  const int n0 = (blockIdx.x * 4 + wave) * 64;
  const int m0 = blockIdx.y * 64;
  const int b  = blockIdx.z;

  const _Float16* A0 = W + (size_t)(m0 + n) * 256;
  const _Float16* B0 = xnT + ((size_t)b * L_TOT + (n0 + n)) * 256;

  v8f acc[4][4] = {};
  for (int kk = 0; kk < 256; kk += 32) {
    v16h af[4], bf[4];
#pragma unroll
    for (int mt = 0; mt < 4; ++mt) af[mt] = load_a_frag(A0 + (size_t)mt * 16 * 256 + kk, hi);
#pragma unroll
    for (int nt = 0; nt < 4; ++nt) bf[nt] = load_b_frag(B0 + (size_t)nt * 16 * 256 + kk, hi);
#pragma unroll
    for (int mt = 0; mt < 4; ++mt)
#pragma unroll
      for (int nt = 0; nt < 4; ++nt)
        acc[mt][nt] = wmma_f16(af[mt], bf[nt], acc[mt][nt]);
  }

#pragma unroll
  for (int mt = 0; mt < 4; ++mt) {
    const int m = m0 + mt * 16;             // 16-row tile never crosses 64 boundary
    const int head = m / 192, seg = m % 192;
    const int bh = b * NH + head;
#pragma unroll
    for (int nt = 0; nt < 4; ++nt) {
      const int l = n0 + nt * 16 + n;
      if (seg < 128) {                      // q or k -> transposed contiguous store
        const bool isq = seg < 64;
        const int cbase = isq ? seg : seg - 64;
        v8h outv;
#pragma unroll
        for (int v = 0; v < 8; ++v)
          outv[v] = (_Float16)((acc[mt][nt][v] + bias[m + v + 8 * hi]) * QKV_SCALE);
        _Float16* dst = (isq ? qT : kT) + ((size_t)bh * L_TOT + l) * CH + cbase + 8 * hi;
        *(v8h*)dst = outv;
      } else {                              // v -> row-major (BH,64,L)
        const int cbase = seg - 128;
#pragma unroll
        for (int v = 0; v < 8; ++v)
          vmat[((size_t)bh * CH + (cbase + v + 8 * hi)) * L_TOT + l] =
              (_Float16)(acc[mt][nt][v] + bias[m + v + 8 * hi]);
      }
    }
  }
}

// ---------------------------------------------------------- Flash attention (WMMA)
// 256 threads = 8 waves, one 16-row t-tile each, all sharing the same (bh)
// K/V stream: K(32x64) and V(64x32) tiles double-buffered in LDS.
#define ATT_WAVES 8
__global__ __launch_bounds__(256) void attention_kernel(
    const _Float16* __restrict__ qT,   // (BH,L,64) pre-scaled
    const _Float16* __restrict__ kT,   // (BH,L,64) pre-scaled
    const _Float16* __restrict__ vmat, // (BH,64,L)
    _Float16* __restrict__ aT)         // (B,L,256)
{
  // padded rows: K 72 halves (144B), V/P 40 halves (80B) -> 16B aligned frags
  __shared__ __align__(16) _Float16 klds[2][32][72];
  __shared__ __align__(16) _Float16 vlds[2][64][40];
  __shared__ __align__(16) _Float16 plds[ATT_WAVES][16][40];

  const int tid = threadIdx.x;
  const int wave = tid >> 5, lane = tid & 31;
  const int n = lane & 15, hi = lane >> 4;
  const int bh = blockIdx.y;
  const int t0 = (blockIdx.x * ATT_WAVES + wave) * 16;

  const _Float16* qbase = qT + (size_t)bh * L_TOT * CH;
  const _Float16* kbase = kT + (size_t)bh * L_TOT * CH;
  const _Float16* vbase = vmat + (size_t)bh * CH * L_TOT;

  // cooperative staging map: 256 threads x 16B per tensor
  const int krow = tid >> 3, kcol = (tid & 7) * 8;   // 32 rows x 64 halves
  const int vrow = tid >> 2, vcol = (tid & 3) * 8;   // 64 rows x 32 halves

  // Q fragments for this t-tile (K=64 -> two 32-wide steps), reused for all s
  const v16h qa0 = load_a_frag(qbase + (size_t)(t0 + n) * CH + 0, hi);
  const v16h qa1 = load_a_frag(qbase + (size_t)(t0 + n) * CH + 32, hi);

  v8f acc[4] = {};
  float mrow[8], lrow[8];
#pragma unroll
  for (int v = 0; v < 8; ++v) { mrow[v] = -1e30f; lrow[v] = 0.f; }

  // prologue: stage s-block 0 into buffer 0
#if HAVE_ASYNC_LDS
  async_copy16(kbase + (size_t)krow * CH + kcol, &klds[0][krow][kcol]);
  async_copy16(vbase + (size_t)vrow * L_TOT + vcol, &vlds[0][vrow][vcol]);
  wait_async0();
#else
  {
    v8h kr = *(const v8h*)(kbase + (size_t)krow * CH + kcol);
    v8h vr = *(const v8h*)(vbase + (size_t)vrow * L_TOT + vcol);
    *(v8h*)(&klds[0][krow][kcol]) = kr;
    *(v8h*)(&vlds[0][vrow][vcol]) = vr;
  }
#endif
  __syncthreads();

  int cur = 0;
  for (int s0 = 0; s0 < L_TOT; s0 += 32) {
    const bool have_next = (s0 + 32) < L_TOT;
#if HAVE_ASYNC_LDS
    // issue async stage of next block into the free buffer; it lands any time
    // before the wait below. Free since all reads of it finished last iter.
    if (have_next) {
      async_copy16(kbase + (size_t)(s0 + 32 + krow) * CH + kcol, &klds[cur ^ 1][krow][kcol]);
      async_copy16(vbase + (size_t)vrow * L_TOT + (s0 + 32) + vcol, &vlds[cur ^ 1][vrow][vcol]);
    }
#else
    v8h knext = {}, vnext = {};
    if (have_next) {
      knext = *(const v8h*)(kbase + (size_t)(s0 + 32 + krow) * CH + kcol);
      vnext = *(const v8h*)(vbase + (size_t)vrow * L_TOT + (s0 + 32) + vcol);
    }
#endif
    if (s0 + 64 < L_TOT) {   // hint block-after-next into cache
      __builtin_prefetch(kbase + (size_t)(s0 + 64 + krow) * CH + kcol, 0, 1);
      __builtin_prefetch(vbase + (size_t)vrow * L_TOT + (s0 + 64) + vcol, 0, 1);
    }

    // ---- S = Qt x K : two 16x16 column tiles, B-frags from LDS
    v8f sA = {}, sB = {};
    sA = wmma_f16(qa0, load_b_frag(&klds[cur][n][0],       hi), sA);
    sB = wmma_f16(qa0, load_b_frag(&klds[cur][16 + n][0],  hi), sB);
    sA = wmma_f16(qa1, load_b_frag(&klds[cur][n][32],      hi), sA);
    sB = wmma_f16(qa1, load_b_frag(&klds[cur][16 + n][32], hi), sB);

    // ---- online softmax over 32 new columns (rows live in half-waves)
    _Float16 (*pt)[40] = plds[wave];
#pragma unroll
    for (int v = 0; v < 8; ++v) {
      float mx = fmaxf(sA[v], sB[v]);
#pragma unroll
      for (int off = 8; off > 0; off >>= 1) mx = fmaxf(mx, __shfl_xor(mx, off, 32));
      float mnew = fmaxf(mrow[v], mx);
      float alpha = __expf(mrow[v] - mnew);
      float p0 = __expf(sA[v] - mnew);
      float p1 = __expf(sB[v] - mnew);
      float ps = p0 + p1;
#pragma unroll
      for (int off = 8; off > 0; off >>= 1) ps += __shfl_xor(ps, off, 32);
      lrow[v] = lrow[v] * alpha + ps;
      mrow[v] = mnew;
      acc[0][v] *= alpha; acc[1][v] *= alpha; acc[2][v] *= alpha; acc[3][v] *= alpha;
      int r = v + 8 * hi;
      pt[r][n]      = (_Float16)p0;
      pt[r][16 + n] = (_Float16)p1;
    }
    __syncthreads();   // P visible; all reads of buf[cur^1] (prev iter) done

#if !HAVE_ASYNC_LDS
    // stage next K/V into the free buffer while we finish this block
    if (have_next) {
      *(v8h*)(&klds[cur ^ 1][krow][kcol]) = knext;
      *(v8h*)(&vlds[cur ^ 1][vrow][vcol]) = vnext;
    }
#endif

    // P tile: C-layout -> A-layout via LDS
    v16h pa;
    {
      v8h lo = *(const v8h*)(&pt[n][hi * 8]);
      v8h hh = *(const v8h*)(&pt[n][16 + hi * 8]);
      pa = combine16(lo, hh);
    }

    // ---- O += P x V (K = 32 s-columns, 4 c-tiles of 16)
#pragma unroll
    for (int ct = 0; ct < 4; ++ct)
      acc[ct] = wmma_f16(pa, load_b_frag(&vlds[cur][ct * 16 + n][0], hi), acc[ct]);

#if HAVE_ASYNC_LDS
    if (have_next) wait_async0();   // next buffer fully landed in LDS
#endif
    __syncthreads();   // next-buffer writes done; P/V reads done
    cur ^= 1;
  }

  // ---- normalize and write transposed output aT (B,L,256)
  const int b = bh >> 2, head = bh & 3;
#pragma unroll
  for (int ct = 0; ct < 4; ++ct) {
    int gc = head * CH + ct * 16 + n;
#pragma unroll
    for (int v = 0; v < 8; ++v) {
      int t = t0 + v + 8 * hi;
      aT[((size_t)b * L_TOT + t) * C_TOT + gc] = (_Float16)(acc[ct][v] / lrow[v]);
    }
  }
}

// ---------------------------------------------------------------- proj GEMM (WMMA)
// 64x64 tile per wave, epilogue adds bias + fp32 residual.
__global__ __launch_bounds__(128) void proj_gemm_kernel(
    const _Float16* __restrict__ W,     // 256x256
    const _Float16* __restrict__ aT,    // (B,L,256)
    const float* __restrict__ bias,     // 256
    const float* __restrict__ xres,     // (B,256,L)
    float* __restrict__ out)            // (B,256,L)
{
  const int wave = threadIdx.x >> 5, lane = threadIdx.x & 31;
  const int n = lane & 15, hi = lane >> 4;
  const int n0 = (blockIdx.x * 4 + wave) * 64;
  const int m0 = blockIdx.y * 64;
  const int b  = blockIdx.z;

  const _Float16* A0 = W + (size_t)(m0 + n) * 256;
  const _Float16* B0 = aT + ((size_t)b * L_TOT + (n0 + n)) * 256;

  v8f acc[4][4] = {};
  for (int kk = 0; kk < 256; kk += 32) {
    v16h af[4], bf[4];
#pragma unroll
    for (int mt = 0; mt < 4; ++mt) af[mt] = load_a_frag(A0 + (size_t)mt * 16 * 256 + kk, hi);
#pragma unroll
    for (int nt = 0; nt < 4; ++nt) bf[nt] = load_b_frag(B0 + (size_t)nt * 16 * 256 + kk, hi);
#pragma unroll
    for (int mt = 0; mt < 4; ++mt)
#pragma unroll
      for (int nt = 0; nt < 4; ++nt)
        acc[mt][nt] = wmma_f16(af[mt], bf[nt], acc[mt][nt]);
  }

#pragma unroll
  for (int mt = 0; mt < 4; ++mt)
#pragma unroll
    for (int nt = 0; nt < 4; ++nt)
#pragma unroll
      for (int v = 0; v < 8; ++v) {
        int o = m0 + mt * 16 + v + 8 * hi;
        size_t idx = ((size_t)b * C_TOT + o) * L_TOT + n0 + nt * 16 + n;
        out[idx] = acc[mt][nt][v] + bias[o] + xres[idx];
      }
}

// ---------------------------------------------------------------------------
extern "C" void kernel_launch(void* const* d_in, const int* in_sizes, int n_in,
                              void* d_out, int out_size, void* d_ws, size_t ws_size,
                              hipStream_t stream) {
  (void)in_sizes; (void)n_in; (void)out_size; (void)ws_size;
  const float* x      = (const float*)d_in[0];
  const float* gn_w   = (const float*)d_in[1];
  const float* gn_b   = (const float*)d_in[2];
  const float* qkv_w  = (const float*)d_in[3];
  const float* qkv_b  = (const float*)d_in[4];
  const float* proj_w = (const float*)d_in[5];
  const float* proj_b = (const float*)d_in[6];
  float* out = (float*)d_out;

  char* ws = (char*)d_ws;
  size_t off = 0;
  auto salloc = [&](size_t bytes) -> void* {
    void* p = ws + off;
    off = (off + bytes + 255) & ~(size_t)255;
    return p;
  };
  float*    mean    = (float*)salloc(128 * sizeof(float));
  float*    rstd    = (float*)salloc(128 * sizeof(float));
  _Float16* qkvw_h  = (_Float16*)salloc((size_t)768 * 256 * 2);
  _Float16* projw_h = (_Float16*)salloc((size_t)256 * 256 * 2);
  _Float16* xnT     = (_Float16*)salloc((size_t)B_TOT * L_TOT * C_TOT * 2);   // 8 MB
  _Float16* qT      = (_Float16*)salloc((size_t)BH_TOT * L_TOT * CH * 2);     // 8 MB
  _Float16* kTm     = (_Float16*)salloc((size_t)BH_TOT * L_TOT * CH * 2);     // 8 MB
  _Float16* vm      = (_Float16*)salloc((size_t)BH_TOT * CH * L_TOT * 2);     // 8 MB
  _Float16* aT      = (_Float16*)salloc((size_t)B_TOT * L_TOT * C_TOT * 2);   // 8 MB
  // total ws use: ~40.5 MB

  f32_to_f16_kernel<<<768, 256, 0, stream>>>(qkv_w, qkvw_h, 768 * 256);
  f32_to_f16_kernel<<<256, 256, 0, stream>>>(proj_w, projw_h, 256 * 256);

  gn_stats_kernel<<<128, 256, 0, stream>>>(x, mean, rstd);
  gn_apply_t_kernel<<<dim3(L_TOT / 32, C_TOT / 32, B_TOT), dim3(32, 8), 0, stream>>>(
      x, gn_w, gn_b, mean, rstd, xnT);

  qkv_gemm_kernel<<<dim3(L_TOT / 256, 768 / 64, B_TOT), 128, 0, stream>>>(
      qkvw_h, xnT, qkv_b, qT, kTm, vm);

  attention_kernel<<<dim3(L_TOT / (16 * ATT_WAVES), BH_TOT), 256, 0, stream>>>(
      qT, kTm, vm, aT);

  proj_gemm_kernel<<<dim3(L_TOT / 256, C_TOT / 64, B_TOT), 128, 0, stream>>>(
      projw_h, aT, proj_b, x, out);
}
